// RPN_3719441679033
// MI455X (gfx1250) — matrix operand order repaired and checked by reading, not compiled, not run on hardware
//
#include <hip/hip_runtime.h>
#include <math.h>

typedef __bf16 v16bf __attribute__((ext_vector_type(16)));
typedef float  v8f   __attribute__((ext_vector_type(8)));

#define C_IN 256
#define C_OUT 256
#define NBATCH 2
#define NTOT 261888            // sum over levels of 3*H*W
#define PRE_NMS 1000
#define IOU_THRESH 0.7f
#define SCALE_CLAMP 4.135166556742356f   // log(1000/16)
#define IM_SZ 1024.0f

// ---------------- helpers ----------------
__device__ __forceinline__ unsigned short f2bf(float f) {
  unsigned u = __float_as_uint(f);
  unsigned r = u + 0x7FFFu + ((u >> 16) & 1u);
  return (unsigned short)(r >> 16);
}
__device__ __forceinline__ float bf2f(unsigned short h) {
  return __uint_as_float(((unsigned)h) << 16);
}
__device__ __forceinline__ unsigned keyof(float f) {
  unsigned u = __float_as_uint(f);
  return u ^ ((u & 0x80000000u) ? 0xFFFFFFFFu : 0x80000000u);
}

union Frag16 { uint4 q[2]; v16bf v; };   // 16 bf16 = 32 B

// ---------------- kernel 0: pack conv weights to bf16 hi/lo ----------------
// dst layout: [cb(8)][tap(9)][co(256)][kk(32)]
__global__ void rpn_pack_w(const float* __restrict__ conv_w,
                           unsigned short* __restrict__ wph,
                           unsigned short* __restrict__ wpl) {
  int idx = blockIdx.x * 256 + threadIdx.x;
  if (idx >= 8 * 9 * 32 * 256) return;
  int kk = idx & 31;
  int k3 = idx >> 5;
  int co = k3 & 255;
  int t2 = k3 >> 8;
  int tap = t2 % 9;
  int cb  = t2 / 9;
  int ci  = cb * 32 + kk;
  float v = conv_w[((size_t)co * C_IN + ci) * 9 + tap];
  unsigned short h = f2bf(v);
  wph[idx] = h;
  wpl[idx] = f2bf(v - bf2f(h));
}

// ---------------- kernel 1: fused conv3x3+relu + heads + decode ----------------
// LDS: input tile staged as bf16 hi/lo, channel-fastest so an A-fragment is
// two contiguous 16B runs -> ds_load_b128; conversion done ONCE at staging.
struct SmemConv {
  union {
    struct {
      alignas(16) unsigned short hi[3][34][32];  // [row][col][kk]  6528 B
      alignas(16) unsigned short lo[3][34][32];  //                 6528 B
    } s;
    float t[32][260];                            // [pos][co]      33280 B
  } u;
  float dvals[32][12];
};

__global__ __launch_bounds__(256)
void rpn_conv_heads(const float* __restrict__ x,
                    const unsigned short* __restrict__ wph,
                    const unsigned short* __restrict__ wpl,
                    const float* __restrict__ conv_b,
                    const float* __restrict__ obj_w,
                    const float* __restrict__ obj_b,
                    const float* __restrict__ delta_w,
                    const float* __restrict__ delta_b,
                    float* __restrict__ scores,
                    float* __restrict__ boxes,
                    int H, int W, float stride_, float asize, int lvl_off) {
  __shared__ SmemConv sm;

  const int tid  = threadIdx.x;
  const int lane = tid & 31;
  const int wave = tid >> 5;
  const int mh   = wave & 1;       // which 16-row half of the 32-position tile
  const int nq   = wave >> 1;      // which group of 4 N-tiles (64 channels)

  const int tilesX = (W + 31) >> 5;
  int b   = blockIdx.x;
  int tx  = b % tilesX;
  int rem = b / tilesX;
  int y   = rem % H;
  int n   = rem / H;
  int x0  = tx << 5;

  v8f acc[4] = {};

  const int mrow   = lane & 15;
  const int hihalf = lane >> 4;
  const int xbase  = mh * 16 + mrow;

  for (int cb = 0; cb < 8; ++cb) {
    __syncthreads();
    // stage input slice: 32 ch x 3 rows x 34 cols, zero padded, f32->bf16 hi/lo
    // (global reads coalesced along cols; transposed store to channel-fastest LDS)
    for (int t = tid; t < 32 * 3 * 34; t += 256) {
      int c   = t % 34;
      int rr  = t / 34;
      int ry  = rr % 3;
      int ciL = rr / 3;
      int iy  = y + ry - 1;
      int ix  = x0 - 1 + c;
      float v = 0.f;
      if (iy >= 0 && iy < H && ix >= 0 && ix < W)
        v = x[(((size_t)n * C_IN + cb * 32 + ciL) * H + iy) * W + ix];
      unsigned short h = f2bf(v);
      sm.u.s.hi[ry][c][ciL] = h;
      sm.u.s.lo[ry][c][ciL] = f2bf(v - bf2f(h));
    }
    __syncthreads();

    if (cb < 7)  // warm L2 for next channel block's weights
      __builtin_prefetch(wph + ((size_t)((cb + 1) * 9) * 32) * 256, 0, 3);

#pragma unroll
    for (int tap = 0; tap < 9; ++tap) {
      const int ky = tap / 3, kx = tap % 3;
      // A fragment: two contiguous 8-channel runs per lane -> 2x ds_load_b128 each
      const unsigned short* hp = &sm.u.s.hi[ky][xbase + kx][hihalf * 8];
      const unsigned short* lp = &sm.u.s.lo[ky][xbase + kx][hihalf * 8];
      Frag16 ah, al;
      ah.q[0] = *(const uint4*)(hp);
      ah.q[1] = *(const uint4*)(hp + 16);   // kk runs [8*hh..8*hh+8) and [16+8*hh..)
      al.q[0] = *(const uint4*)(lp);
      al.q[1] = *(const uint4*)(lp + 16);

      const size_t kbase = ((size_t)(cb * 9 + tap) * 256 + (lane & 15)) * 32 + hihalf * 8;
#pragma unroll
      for (int j = 0; j < 4; ++j) {
        int co_base = (nq * 4 + j) * 16;
        const uint4* ph = (const uint4*)(wph + kbase + (size_t)co_base * 32);
        const uint4* pl = (const uint4*)(wpl + kbase + (size_t)co_base * 32);
        Frag16 bh, bl;
        bh.q[0] = ph[0]; bh.q[1] = ph[2];
        bl.q[0] = pl[0]; bl.q[1] = pl[2];
        acc[j] = __builtin_amdgcn_wmma_f32_16x16x32_bf16(false, ah.v, false, bh.v,
                                                         (short)0, acc[j], false, false);
        acc[j] = __builtin_amdgcn_wmma_f32_16x16x32_bf16(false, ah.v, false, bl.v,
                                                         (short)0, acc[j], false, false);
        acc[j] = __builtin_amdgcn_wmma_f32_16x16x32_bf16(false, al.v, false, bh.v,
                                                         (short)0, acc[j], false, false);
      }
    }
  }
  __syncthreads();   // done reading staged input; reuse LDS for t

  // bias + relu, spill t tile to LDS: C/D layout M = r + 8*(lane>=16), N = lane&15
#pragma unroll
  for (int j = 0; j < 4; ++j) {
    int co = (nq * 4 + j) * 16 + (lane & 15);
    float bias = conv_b[co];
#pragma unroll
    for (int r = 0; r < 8; ++r) {
      int pos = mh * 16 + r + ((lane >> 4) << 3);
      float v = acc[j][r] + bias;
      sm.u.t[pos][co] = v > 0.f ? v : 0.f;
    }
  }
  __syncthreads();

  // 1x1 heads: 32 positions x (3 obj + 12 delta) dots of length 256
  for (int t = tid; t < 32 * 15; t += 256) {
    int pos = t / 15, oc = t % 15;
    const float* wr = (oc < 3) ? (obj_w + (size_t)oc * C_OUT)
                               : (delta_w + (size_t)(oc - 3) * C_OUT);
    float s = 0.f;
#pragma unroll 4
    for (int c = 0; c < C_OUT; ++c) s += sm.u.t[pos][c] * wr[c];
    if (oc < 3) {
      int xg = x0 + pos;
      if (xg < W) {
        float sc = 1.f / (1.f + expf(-(s + obj_b[oc])));
        scores[(size_t)n * NTOT + lvl_off + (size_t)(y * W + xg) * 3 + oc] = sc;
      }
    } else {
      sm.dvals[pos][oc - 3] = s + delta_b[oc - 3];
    }
  }
  __syncthreads();

  // decode + clip (32 positions x 3 anchors)
  if (tid < 96) {
    int pos = tid / 3, a = tid % 3;
    int xg = x0 + pos;
    if (xg < W) {
      float r  = (a == 0) ? 0.5f : (a == 1) ? 1.f : 2.f;
      float wa = sqrtf(asize * asize / r);
      float ha = wa * r;
      float cxa = (float)xg * stride_;
      float cya = (float)y * stride_;
      float dx = sm.dvals[pos][a * 4 + 0];
      float dy = sm.dvals[pos][a * 4 + 1];
      float dw = fminf(sm.dvals[pos][a * 4 + 2], SCALE_CLAMP);
      float dh = fminf(sm.dvals[pos][a * 4 + 3], SCALE_CLAMP);
      float cx = dx * wa + cxa;
      float cy = dy * ha + cya;
      float bw = expf(dw) * wa;
      float bh = expf(dh) * ha;
      size_t bi = ((size_t)n * NTOT + lvl_off + (size_t)(y * W + xg) * 3 + a) * 4;
      boxes[bi + 0] = fminf(fmaxf(cx - 0.5f * bw, 0.f), IM_SZ);
      boxes[bi + 1] = fminf(fmaxf(cy - 0.5f * bh, 0.f), IM_SZ);
      boxes[bi + 2] = fminf(fmaxf(cx + 0.5f * bw, 0.f), IM_SZ);
      boxes[bi + 3] = fminf(fmaxf(cy + 0.5f * bh, 0.f), IM_SZ);
    }
  }
}

// ---------------- kernel 2: exact global top-1000 per image ----------------
__global__ __launch_bounds__(1024)
void rpn_topk(const float* __restrict__ scores,
              const float* __restrict__ boxes,
              float* __restrict__ tkbox) {
  __shared__ unsigned hist[256];
  __shared__ unsigned skey[1024];
  __shared__ int      sidx[1024];
  __shared__ unsigned s_sel, s_kneed, s_cnt;

  const int tid = threadIdx.x;
  const int n   = blockIdx.x;
  const float* sc = scores + (size_t)n * NTOT;

  unsigned prefix = 0, pmask = 0, kneed = PRE_NMS;
  for (int p = 0; p < 4; ++p) {
    int shift = 24 - 8 * p;
    if (tid < 256) hist[tid] = 0;
    __syncthreads();
    for (int i = tid; i < NTOT; i += 1024) {
      unsigned k = keyof(sc[i]);
      if ((k & pmask) == prefix) atomicAdd(&hist[(k >> shift) & 255u], 1u);
    }
    __syncthreads();
    if (tid == 0) {
      unsigned cum = 0;
      for (int b = 255; b >= 0; --b) {
        cum += hist[b];
        if (cum >= kneed) { s_sel = (unsigned)b; s_kneed = kneed - (cum - hist[b]); break; }
      }
    }
    __syncthreads();
    prefix |= s_sel << shift;
    pmask  |= 0xFFu << shift;
    kneed   = s_kneed;
    __syncthreads();
  }
  const unsigned T = prefix;
  const unsigned needed_eq = kneed;

  // deterministic tie-break: smallest-index elements among key==T
  int lo = 0, hi = NTOT;
  while (lo < hi) {
    int mid = (lo + hi) >> 1;
    if (tid == 0) s_cnt = 0;
    __syncthreads();
    unsigned c = 0;
    for (int i = tid; i < mid; i += 1024)
      if (keyof(sc[i]) == T) c++;
    atomicAdd(&s_cnt, c);
    __syncthreads();
    unsigned tot = s_cnt;
    __syncthreads();
    if (tot >= needed_eq) hi = mid; else lo = mid + 1;
  }
  const int ithr = lo;

  if (tid == 0) s_cnt = 0;
  __syncthreads();
  for (int i = tid; i < NTOT; i += 1024) {
    unsigned k = keyof(sc[i]);
    if (k > T || (k == T && i < ithr)) {
      unsigned p = atomicAdd(&s_cnt, 1u);
      skey[p] = k; sidx[p] = i;
    }
  }
  __syncthreads();
  if (tid >= PRE_NMS) { skey[tid] = 0u; sidx[tid] = 0x7FFFFFFF; }
  __syncthreads();

  // bitonic sort 1024: key desc, idx asc (deterministic total order)
  for (unsigned k = 2; k <= 1024; k <<= 1) {
    for (unsigned j = k >> 1; j > 0; j >>= 1) {
      __syncthreads();
      unsigned i = (unsigned)tid, ixj = i ^ j;
      if (ixj > i) {
        unsigned ak = skey[i], bk = skey[ixj];
        int ai = sidx[i], bi = sidx[ixj];
        bool aFirst = (ak > bk) || (ak == bk && ai < bi);
        bool up = ((i & k) == 0);
        if (up ? !aFirst : aFirst) {
          skey[i] = bk; skey[ixj] = ak;
          sidx[i] = bi; sidx[ixj] = ai;
        }
      }
    }
  }
  __syncthreads();

  if (tid < PRE_NMS) {
    int idx = sidx[tid];
    size_t src = ((size_t)n * NTOT + idx) * 4;
    size_t dst = ((size_t)n * PRE_NMS + tid) * 4;
    tkbox[dst + 0] = boxes[src + 0];
    tkbox[dst + 1] = boxes[src + 1];
    tkbox[dst + 2] = boxes[src + 2];
    tkbox[dst + 3] = boxes[src + 3];
  }
}

// ---------------- kernel 3: NMS + stable partition + output ----------------
__global__ __launch_bounds__(1024)
void rpn_nms(const float* __restrict__ tkbox, float* __restrict__ out) {
  __shared__ float bx1[1024], by1[1024], bx2[1024], by2[1024];
  __shared__ int   keep[1024];
  __shared__ int   scan[1024];

  const int tid = threadIdx.x;
  const int n   = blockIdx.x;

  if (tid < PRE_NMS) {
    size_t b = ((size_t)n * PRE_NMS + tid) * 4;
    bx1[tid] = tkbox[b + 0]; by1[tid] = tkbox[b + 1];
    bx2[tid] = tkbox[b + 2]; by2[tid] = tkbox[b + 3];
    keep[tid] = 1;                   // MIN_SIZE==0 and clipped boxes => valid
  } else {
    keep[tid] = 0;
  }
  __syncthreads();

  float myArea = (bx2[tid] - bx1[tid]) * (by2[tid] - by1[tid]);
  for (int i = 0; i < PRE_NMS; ++i) {
    int ki = keep[i];
    if (ki && tid > i && tid < PRE_NMS && keep[tid]) {
      float ix1 = fmaxf(bx1[i], bx1[tid]);
      float iy1 = fmaxf(by1[i], by1[tid]);
      float ix2 = fminf(bx2[i], bx2[tid]);
      float iy2 = fminf(by2[i], by2[tid]);
      float iw = ix2 - ix1; if (iw < 0.f) iw = 0.f;
      float ih = iy2 - iy1; if (ih < 0.f) ih = 0.f;
      float inter = iw * ih;
      float areai = (bx2[i] - bx1[i]) * (by2[i] - by1[i]);
      float uni = fmaxf(areai + myArea - inter, 1e-6f);
      if (inter / uni > IOU_THRESH) keep[tid] = 0;
    }
    __syncthreads();
  }

  // inclusive scan of keep (Hillis-Steele)
  int v = keep[tid];
  scan[tid] = v;
  __syncthreads();
  for (int off = 1; off < 1024; off <<= 1) {
    int add = (tid >= off) ? scan[tid - off] : 0;
    __syncthreads();
    scan[tid] += add;
    __syncthreads();
  }
  int rank = scan[tid] - v;   // exclusive rank among kept

  // zero output rows for this image (ret: N*1000 x 5, then dets: N*1000 x 4)
  float* ret  = out;
  float* dets = out + (size_t)NBATCH * PRE_NMS * 5;
  if (tid < PRE_NMS) {
    size_t rr = ((size_t)n * PRE_NMS + tid);
    for (int c = 0; c < 5; ++c) ret[rr * 5 + c] = 0.f;
    for (int c = 0; c < 4; ++c) dets[rr * 4 + c] = 0.f;
  }
  __syncthreads();
  if (tid < PRE_NMS && v) {
    size_t rr = ((size_t)n * PRE_NMS + rank);
    ret[rr * 5 + 1] = bx1[tid];
    ret[rr * 5 + 2] = by1[tid];
    ret[rr * 5 + 3] = bx2[tid];
    ret[rr * 5 + 4] = by2[tid];
    dets[rr * 4 + 0] = bx1[tid];
    dets[rr * 4 + 1] = by1[tid];
    dets[rr * 4 + 2] = bx2[tid];
    dets[rr * 4 + 3] = by2[tid];
  }
}

// ---------------- host launcher ----------------
extern "C" void kernel_launch(void* const* d_in, const int* in_sizes, int n_in,
                              void* d_out, int out_size, void* d_ws, size_t ws_size,
                              hipStream_t stream) {
  (void)in_sizes; (void)n_in; (void)out_size; (void)ws_size;
  const float* feats[5] = {(const float*)d_in[0], (const float*)d_in[1],
                           (const float*)d_in[2], (const float*)d_in[3],
                           (const float*)d_in[4]};
  const float* conv_w  = (const float*)d_in[5];
  const float* conv_b  = (const float*)d_in[6];
  const float* obj_w   = (const float*)d_in[7];
  const float* obj_b   = (const float*)d_in[8];
  const float* delta_w = (const float*)d_in[9];
  const float* delta_b = (const float*)d_in[10];
  float* out = (float*)d_out;

  char* ws = (char*)d_ws;
  const size_t WPK_ELEMS = (size_t)8 * 9 * 32 * 256;     // 589824
  unsigned short* wph = (unsigned short*)(ws);
  unsigned short* wpl = (unsigned short*)(ws + WPK_ELEMS * 2);
  float* scores = (float*)(ws + WPK_ELEMS * 4);                            // 2*NTOT
  float* boxes  = (float*)(ws + WPK_ELEMS * 4 + (size_t)NBATCH * NTOT * 4);// 2*NTOT*4
  float* tkbox  = (float*)(ws + WPK_ELEMS * 4 + (size_t)NBATCH * NTOT * 4
                              + (size_t)NBATCH * NTOT * 16);               // 2*1000*4

  rpn_pack_w<<<(int)((WPK_ELEMS + 255) / 256), 256, 0, stream>>>(conv_w, wph, wpl);

  const int   Hs[5]      = {256, 128, 64, 32, 16};
  const float strides[5] = {4.f, 8.f, 16.f, 32.f, 64.f};
  const float sizes_[5]  = {32.f, 64.f, 128.f, 256.f, 512.f};
  const int   offs[5]    = {0, 196608, 245760, 258048, 261120};

  for (int lvl = 0; lvl < 5; ++lvl) {
    int H = Hs[lvl], W = Hs[lvl];
    int tilesX = (W + 31) / 32;
    int grid = NBATCH * H * tilesX;
    rpn_conv_heads<<<grid, 256, 0, stream>>>(feats[lvl], wph, wpl, conv_b,
                                             obj_w, obj_b, delta_w, delta_b,
                                             scores, boxes,
                                             H, W, strides[lvl], sizes_[lvl], offs[lvl]);
  }

  rpn_topk<<<NBATCH, 1024, 0, stream>>>(scores, boxes, tkbox);
  rpn_nms<<<NBATCH, 1024, 0, stream>>>(tkbox, out);
}
